// PN2GeometryEncoderMSG_6734508720337
// MI455X (gfx1250) — compile-verified, matrix operand order checked by reading, not run on hardware
//
#include <hip/hip_runtime.h>
#include <hip/hip_bf16.h>

typedef __attribute__((ext_vector_type(16))) _Float16 v16h;
typedef __attribute__((ext_vector_type(8)))  _Float16 v8h;
typedef __attribute__((ext_vector_type(8)))  float    v8f;

// ---------------------------------------------------------------------------
// Fused GEMM + BN + ReLU:  Y[M,Cout] = relu(alpha .* (X[M,Kp] @ Wh[Cout,Kp]^T) + beta)
// X, Wh are f16 row-major with Kp a multiple of 32. One wave = one 16x16 tile.
// OOB A-rows are clamped to row M-1 (their D rows are computed but never
// stored), which keeps the K loop completely branch-free and EXEC-uniform.
// ---------------------------------------------------------------------------
__global__ __launch_bounds__(256) void k_gemm_bn_relu(
    const _Float16* __restrict__ X, const _Float16* __restrict__ W,
    const float* __restrict__ alpha, const float* __restrict__ beta,
    _Float16* __restrict__ Y, int M, int Kp, int Cout)
{
    const int lane = threadIdx.x & 31;
    const int wave = threadIdx.x >> 5;
    const int tilesN = Cout >> 4;
    const int tilesM = (M + 15) >> 4;
    const int tile = blockIdx.x * (blockDim.x >> 5) + wave;
    if (tile >= tilesM * tilesN) return;
    const int tm = tile / tilesN;
    const int tn = tile - tm * tilesN;

    const int r16 = lane & 15;            // A row / D column within tile
    const int kb  = (lane >> 4) << 3;     // 0 or 8: K sub-block for this lane half
    int arow = tm * 16 + r16;
    if (arow >= M) arow = M - 1;          // clamp: garbage rows are masked at store
    const _Float16* __restrict__ xp = X + (size_t)arow * Kp + kb;
    const _Float16* __restrict__ wp = W + (size_t)(tn * 16 + r16) * Kp + kb;

    auto step = [&](int kk, v8f c) -> v8f {
        v8h alo = *(const v8h*)(xp + kk);        // K = kk+kb .. +7
        v8h ahi = *(const v8h*)(xp + kk + 16);   // K = kk+16+kb .. +7
        v8h blo = *(const v8h*)(wp + kk);
        v8h bhi = *(const v8h*)(wp + kk + 16);
        v16h a, b;
        #pragma unroll
        for (int i = 0; i < 8; ++i) {
            a[i] = alo[i]; a[i + 8] = ahi[i];
            b[i] = blo[i]; b[i + 8] = bhi[i];
        }
        return __builtin_amdgcn_wmma_f32_16x16x32_f16(
            false, a, false, b, (short)0, c, false, false);
    };

    v8f acc = {};
    int k0 = 0;
    for (; k0 + 64 <= Kp; k0 += 64) {
        __builtin_prefetch((const void*)(xp + k0 + 64), 0, 3);  // speculative, safe
        __builtin_prefetch((const void*)(wp + k0 + 64), 0, 3);
        acc = step(k0, acc);
        acc = step(k0 + 32, acc);
    }
    if (k0 < Kp) acc = step(k0, acc);

    const int n = tn * 16 + r16;
    const float al = alpha[n], be = beta[n];
    const int mbase = tm * 16 + ((lane >> 4) << 3);
    #pragma unroll
    for (int v = 0; v < 8; ++v) {
        int m = mbase + v;
        if (m < M) {
            float y = fmaxf(al * acc[v] + be, 0.0f);
            Y[(size_t)m * Cout + n] = (_Float16)y;
        }
    }
}

// ---------------------------------------------------------------------------
// Weight prep: f32 W[Cout,K] -> f16 Wh[Cout,Kp] (zero pad), fold bias+BN.
// ---------------------------------------------------------------------------
__global__ void k_prep(const float* __restrict__ W, const float* __restrict__ bi,
                       const float* __restrict__ ga, const float* __restrict__ bt,
                       const float* __restrict__ rm, const float* __restrict__ rv,
                       _Float16* __restrict__ Wh, float* __restrict__ alpha,
                       float* __restrict__ beta, int Cout, int K, int Kp)
{
    int g = blockIdx.x * blockDim.x + threadIdx.x;
    int total = Cout * Kp;
    if (g < total) {
        int co = g / Kp, kk = g - co * Kp;
        Wh[g] = (kk < K) ? (_Float16)W[(size_t)co * K + kk] : (_Float16)0.f;
    }
    if (g < Cout) {
        float sc = ga[g] * rsqrtf(rv[g] + 1e-5f);
        alpha[g] = sc;
        beta[g]  = (bi[g] - rm[g]) * sc + bt[g];
    }
}

// ---------------------------------------------------------------------------
// Farthest point sampling: one block per batch, sequential (matches reference).
// ---------------------------------------------------------------------------
__global__ __launch_bounds__(256) void k_fps(
    const float* __restrict__ pos, int* __restrict__ out, int N, int nOut)
{
    __shared__ float mind[4096];
    __shared__ float wbm[8];
    __shared__ int   wba[8];
    __shared__ int   sel;
    const int b = blockIdx.x;
    const float* p = pos + (size_t)b * N * 3;
    int* o = out + (size_t)b * nOut;
    const int t = threadIdx.x, nt = blockDim.x;
    const float x0 = p[0], y0 = p[1], z0 = p[2];
    for (int i = t; i < N; i += nt) {
        float dx = p[3*i] - x0, dy = p[3*i+1] - y0, dz = p[3*i+2] - z0;
        mind[i] = dx*dx + dy*dy + dz*dz;
    }
    if (t == 0) o[0] = 0;
    __syncthreads();
    for (int it = 1; it < nOut; ++it) {
        float bm = -1.0f; int ba = N;
        for (int i = t; i < N; i += nt) {
            float v = mind[i];
            if (v > bm || (v == bm && i < ba)) { bm = v; ba = i; }
        }
        for (int off = 16; off > 0; off >>= 1) {
            float ov = __shfl_down(bm, off);
            int   oa = __shfl_down(ba, off);
            if (ov > bm || (ov == bm && oa < ba)) { bm = ov; ba = oa; }
        }
        if ((t & 31) == 0) { wbm[t >> 5] = bm; wba[t >> 5] = ba; }
        __syncthreads();
        if (t == 0) {
            float fb = wbm[0]; int fa = wba[0];
            for (int w = 1; w < (nt >> 5); ++w)
                if (wbm[w] > fb || (wbm[w] == fb && wba[w] < fa)) { fb = wbm[w]; fa = wba[w]; }
            sel = fa; o[it] = fa;
        }
        __syncthreads();
        const int s = sel;
        const float sx = p[3*s], sy = p[3*s+1], sz = p[3*s+2];
        for (int i = t; i < N; i += nt) {
            float dx = p[3*i] - sx, dy = p[3*i+1] - sy, dz = p[3*i+2] - sz;
            float d2 = dx*dx + dy*dy + dz*dz;
            if (d2 < mind[i]) mind[i] = d2;
        }
        __syncthreads();
    }
}

__global__ void k_gather3(const float* __restrict__ pos, const int* __restrict__ idx,
                          float* __restrict__ out, int Npts, int S, int total)
{
    int g = blockIdx.x * blockDim.x + threadIdx.x;
    if (g >= total) return;
    int b = g / S;
    int id = idx[g];
    const float* p = pos + ((size_t)b * Npts + id) * 3;
    out[3*g] = p[0]; out[3*g+1] = p[1]; out[3*g+2] = p[2];
}

// First-k-by-index radius neighbors (matches reference's top_k(-idx) scan order).
__global__ void k_radius(const float* __restrict__ pos, const float* __restrict__ posq,
                         int* __restrict__ nidx, int* __restrict__ ncnt,
                         int N, int S, int k, float r2, int total)
{
    int g = blockIdx.x * blockDim.x + threadIdx.x;
    if (g >= total) return;
    int b = g / S;
    const float* p = pos + (size_t)b * N * 3;
    float qx = posq[3*g], qy = posq[3*g+1], qz = posq[3*g+2];
    int* out = nidx + (size_t)g * k;
    int c = 0;
    for (int j = 0; j < N && c < k; ++j) {
        float dx = p[3*j] - qx, dy = p[3*j+1] - qy, dz = p[3*j+2] - qz;
        if (dx*dx + dy*dy + dz*dz <= r2) out[c++] = j;
    }
    ncnt[g] = c;
    for (int j = c; j < k; ++j) out[j] = 0;
}

// Build grouped MLP input X[(b,s,j), 0:Cf]=x_j, [Cf:Cf+3]=pos_j - pos_i, pad to Kp.
__global__ void k_group(const float* __restrict__ feat, int Cf,
                        const float* __restrict__ pos, const float* __restrict__ posq,
                        const int* __restrict__ nidx, _Float16* __restrict__ X,
                        int N, int S, int k, int Kp, int total)
{
    int g = blockIdx.x * blockDim.x + threadIdx.x;
    if (g >= total) return;
    int bs = g / k;
    int b = bs / S;
    int id = nidx[g];
    const float* f = feat + ((size_t)b * N + id) * Cf;
    _Float16* xr = X + (size_t)g * Kp;
    for (int c = 0; c < Cf; ++c) xr[c] = (_Float16)f[c];
    const float* pj = pos + ((size_t)b * N + id) * 3;
    const float* pq = posq + (size_t)bs * 3;
    xr[Cf]   = (_Float16)(pj[0] - pq[0]);
    xr[Cf+1] = (_Float16)(pj[1] - pq[1]);
    xr[Cf+2] = (_Float16)(pj[2] - pq[2]);
    for (int c = Cf + 3; c < Kp; ++c) xr[c] = (_Float16)0.f;
}

// Masked max over k neighbors (values are post-ReLU >= 0; empty -> 0, as PyG).
__global__ void k_pool_max(const _Float16* __restrict__ Y, const int* __restrict__ ncnt,
                           float* __restrict__ out, int k, int C, int ostride, int ooff,
                           int total)
{
    int g = blockIdx.x * blockDim.x + threadIdx.x;
    if (g >= total) return;
    int c = g % C;
    int bs = g / C;
    int cnt = ncnt[bs];
    const _Float16* y = Y + ((size_t)bs * k) * C + c;
    float m = 0.0f;
    for (int j = 0; j < cnt; ++j) m = fmaxf(m, (float)y[(size_t)j * C]);
    out[(size_t)bs * ostride + ooff + c] = m;
}

__global__ void k_colmax(const float* __restrict__ x, float* __restrict__ out,
                         int S, int C, int total)
{
    int g = blockIdx.x * blockDim.x + threadIdx.x;
    if (g >= total) return;
    int c = g % C, b = g / C;
    const float* xp = x + ((size_t)b * S) * C + c;
    float m = -3.4e38f;
    for (int s = 0; s < S; ++s) m = fmaxf(m, xp[(size_t)s * C]);
    out[(size_t)b * C + c] = m;
}

__global__ void k_f2h_pad(const float* __restrict__ src, _Float16* __restrict__ dst,
                          int C, int Kp, int total)
{
    int g = blockIdx.x * blockDim.x + threadIdx.x;
    if (g >= total) return;
    int c = g % Kp, r = g / Kp;
    dst[g] = (c < C) ? (_Float16)src[(size_t)r * C + c] : (_Float16)0.f;
}

__global__ void k_concat2_f16(const float* __restrict__ a, int Ca,
                              const float* __restrict__ bb, int Cb,
                              _Float16* __restrict__ X, int Kp, int total)
{
    int g = blockIdx.x * blockDim.x + threadIdx.x;
    if (g >= total) return;
    int c = g % Kp, r = g / Kp;
    float v = 0.f;
    if (c < Ca)           v = a[(size_t)r * Ca + c];
    else if (c < Ca + Cb) v = bb[(size_t)r * Cb + (c - Ca)];
    X[g] = (_Float16)v;
}

__global__ void k_knn3(const float* __restrict__ psrc, const float* __restrict__ ptgt,
                       int S, int* __restrict__ kidx, float* __restrict__ kw,
                       int T, int total)
{
    int g = blockIdx.x * blockDim.x + threadIdx.x;   // g = b*T + t
    if (g >= total) return;
    int b = g / T;
    const float* p = psrc + (size_t)b * S * 3;
    float qx = ptgt[3*g], qy = ptgt[3*g+1], qz = ptgt[3*g+2];
    float d0 = 3.4e38f, d1 = 3.4e38f, d2v = 3.4e38f;
    int i0 = 0, i1 = 0, i2 = 0;
    for (int s = 0; s < S; ++s) {
        float dx = p[3*s] - qx, dy = p[3*s+1] - qy, dz = p[3*s+2] - qz;
        float d = dx*dx + dy*dy + dz*dz;
        if (d < d0)       { d2v = d1; i2 = i1; d1 = d0; i1 = i0; d0 = d; i0 = s; }
        else if (d < d1)  { d2v = d1; i2 = i1; d1 = d;  i1 = s; }
        else if (d < d2v) { d2v = d;  i2 = s; }
    }
    kidx[3*g] = i0; kidx[3*g+1] = i1; kidx[3*g+2] = i2;
    kw[3*g]   = 1.0f / fmaxf(d0,  1e-16f);
    kw[3*g+1] = 1.0f / fmaxf(d1,  1e-16f);
    kw[3*g+2] = 1.0f / fmaxf(d2v, 1e-16f);
}

__global__ void k_interp(const float* __restrict__ xsrc, const int* __restrict__ kidx,
                         const float* __restrict__ kw, float* __restrict__ out,
                         int S, int T, int C, int total)
{
    int g = blockIdx.x * blockDim.x + threadIdx.x;
    if (g >= total) return;
    int c = g % C;
    int bt = g / C;
    int b = bt / T;
    const int*   ki = kidx + (size_t)bt * 3;
    const float* w  = kw   + (size_t)bt * 3;
    const float* xb = xsrc + (size_t)b * S * C;
    float num = w[0] * xb[(size_t)ki[0] * C + c]
              + w[1] * xb[(size_t)ki[1] * C + c]
              + w[2] * xb[(size_t)ki[2] * C + c];
    out[g] = num / (w[0] + w[1] + w[2]);
}

__global__ void k_h2f(const _Float16* __restrict__ s, float* __restrict__ d, int total)
{
    int g = blockIdx.x * blockDim.x + threadIdx.x;
    if (g < total) d[g] = (float)s[g];
}

// ---------------------------------------------------------------------------
extern "C" void kernel_launch(void* const* d_in, const int* in_sizes, int n_in,
                              void* d_out, int out_size, void* d_ws, size_t ws_size,
                              hipStream_t stream)
{
    (void)in_sizes; (void)n_in; (void)out_size; (void)ws_size;
    const int B = 8, N = 4096, N1 = 512, N2 = 128;
    const float* pts = (const float*)d_in[0];

    struct LS { int K, Cout; };
    static const LS L[24] = {
        {6,32},{32,32},{32,64},        // sa1 mlp0
        {6,64},{64,64},{64,128},       // sa1 mlp1
        {6,64},{64,96},{96,128},       // sa1 mlp2
        {323,64},{64,64},{64,128},     // sa2 mlp0
        {323,128},{128,128},{128,256}, // sa2 mlp1
        {323,128},{128,128},{128,256}, // sa2 mlp2
        {640,512},{512,256},           // glob
        {960,256},{256,256},           // fp1
        {259,256},{256,256},           // fp0
    };

    size_t off = 0;
    auto carve = [&](size_t bytes) -> void* {
        off = (off + 255) & ~(size_t)255;
        void* p = (char*)d_ws + off;
        off += bytes;
        return p;
    };
    auto cdiv = [](int a, int b) { return (a + b - 1) / b; };

    _Float16* Wh[24]; float* Al[24]; float* Be[24];
    for (int i = 0; i < 24; ++i) {
        int Kp = (L[i].K + 31) & ~31;
        Wh[i] = (_Float16*)carve((size_t)L[i].Cout * Kp * sizeof(_Float16));
        Al[i] = (float*)carve((size_t)L[i].Cout * sizeof(float));
        Be[i] = (float*)carve((size_t)L[i].Cout * sizeof(float));
    }
    _Float16* bufA = (_Float16*)carve((size_t)131072 * 352 * 2); // grouped X (max SA2)
    _Float16* bufB = (_Float16*)carve((size_t)524288 * 128 * 2); // L1/L3 outputs
    _Float16* bufC = (_Float16*)carve((size_t)524288 *  96 * 2); // L2 outputs
    float* x1   = (float*)carve((size_t)B * N1 * 320 * 4);
    float* x2   = (float*)carve((size_t)B * N2 * 640 * 4);
    float* gmax = (float*)carve((size_t)B * 640 * 4);
    float* xup  = (float*)carve((size_t)B * N * 256 * 4);  // also fits B*N1*640
    float* xfp  = (float*)carve((size_t)B * N1 * 256 * 4);
    int*   idx1 = (int*)carve((size_t)B * N1 * 4);
    float* pos1 = (float*)carve((size_t)B * N1 * 3 * 4);
    int*   idx2 = (int*)carve((size_t)B * N2 * 4);
    float* pos2 = (float*)carve((size_t)B * N2 * 3 * 4);
    int*   nidx = (int*)carve((size_t)B * N1 * 128 * 4);
    int*   ncnt = (int*)carve((size_t)B * N1 * 4);
    int*   kidx = (int*)carve((size_t)B * N * 3 * 4);
    float* kw   = (float*)carve((size_t)B * N * 3 * 4);

    // --- weight prep (fold bias + BN, convert to padded f16) ---
    for (int i = 0; i < 24; ++i) {
        int Kp = (L[i].K + 31) & ~31;
        int tot = L[i].Cout * Kp;
        const float* W  = (const float*)d_in[1 + i*6 + 0];
        const float* bi = (const float*)d_in[1 + i*6 + 1];
        const float* ga = (const float*)d_in[1 + i*6 + 2];
        const float* bt = (const float*)d_in[1 + i*6 + 3];
        const float* rm = (const float*)d_in[1 + i*6 + 4];
        const float* rv = (const float*)d_in[1 + i*6 + 5];
        k_prep<<<cdiv(tot, 256), 256, 0, stream>>>(W, bi, ga, bt, rm, rv,
                                                   Wh[i], Al[i], Be[i],
                                                   L[i].Cout, L[i].K, Kp);
    }

    auto gemm = [&](const _Float16* Xp, int li, _Float16* Yp, int M) {
        int Kp = (L[li].K + 31) & ~31, C = L[li].Cout;
        int tiles = ((M + 15) / 16) * (C / 16);
        k_gemm_bn_relu<<<cdiv(tiles, 8), 256, 0, stream>>>(
            Xp, Wh[li], Al[li], Be[li], Yp, M, Kp, C);
    };

    auto sa = [&](const float* feat, int Cf, const float* posAll, int Np,
                  const float* posQ, int S, const float* radii, const int* ks,
                  int li0, float* outF, int ostride, const int* ooffs) {
        for (int sc = 0; sc < 3; ++sc) {
            int k = ks[sc];
            int rows = B * S * k;
            float r2 = radii[sc] * radii[sc];
            k_radius<<<cdiv(B * S, 256), 256, 0, stream>>>(
                posAll, posQ, nidx, ncnt, Np, S, k, r2, B * S);
            int li = li0 + sc * 3;
            int Kp = (L[li].K + 31) & ~31;
            k_group<<<cdiv(rows, 256), 256, 0, stream>>>(
                feat, Cf, posAll, posQ, nidx, bufA, Np, S, k, Kp, rows);
            gemm(bufA, li,     bufB, rows);
            gemm(bufB, li + 1, bufC, rows);
            gemm(bufC, li + 2, bufB, rows);
            int C = L[li + 2].Cout;
            k_pool_max<<<cdiv(B * S * C, 256), 256, 0, stream>>>(
                bufB, ncnt, outF, k, C, ostride, ooffs[sc], B * S * C);
        }
    };

    // --- SA1 ---
    k_fps<<<B, 256, 0, stream>>>(pts, idx1, N, N1);
    k_gather3<<<cdiv(B * N1, 256), 256, 0, stream>>>(pts, idx1, pos1, N, N1, B * N1);
    const float R1[3] = {0.1f, 0.2f, 0.4f};
    const int   K1[3] = {16, 32, 128};
    const int   O1[3] = {0, 64, 192};
    sa(pts, 3, pts, N, pos1, N1, R1, K1, 0, x1, 320, O1);

    // --- SA2 ---
    k_fps<<<B, 256, 0, stream>>>(pos1, idx2, N1, N2);
    k_gather3<<<cdiv(B * N2, 256), 256, 0, stream>>>(pos1, idx2, pos2, N1, N2, B * N2);
    const float R2[3] = {0.2f, 0.4f, 0.8f};
    const int   K2[3] = {32, 64, 128};
    const int   O2[3] = {0, 128, 384};
    sa(x1, 320, pos1, N1, pos2, N2, R2, K2, 9, x2, 640, O2);

    // --- global branch: max over S=128, MLP 640->512->256 ---
    k_colmax<<<cdiv(B * 640, 256), 256, 0, stream>>>(x2, gmax, N2, 640, B * 640);
    k_f2h_pad<<<cdiv(B * 640, 256), 256, 0, stream>>>(gmax, bufA, 640, 640, B * 640);
    gemm(bufA, 18, bufB, B);
    gemm(bufB, 19, bufC, B);
    float* g_out = (float*)d_out + (size_t)B * N * 256;
    k_h2f<<<cdiv(B * 256, 256), 256, 0, stream>>>(bufC, g_out, B * 256);

    // --- FP1: interpolate x2 (pos2 -> pos1), concat x1, MLP 960->256->256 ---
    k_knn3<<<cdiv(B * N1, 256), 256, 0, stream>>>(pos2, pos1, N2, kidx, kw, N1, B * N1);
    k_interp<<<cdiv(B * N1 * 640, 256), 256, 0, stream>>>(
        x2, kidx, kw, xup, N2, N1, 640, B * N1 * 640);
    k_concat2_f16<<<cdiv(B * N1 * 960, 256), 256, 0, stream>>>(
        xup, 640, x1, 320, bufA, 960, B * N1 * 960);
    gemm(bufA, 20, bufB, B * N1);
    gemm(bufB, 21, bufC, B * N1);
    k_h2f<<<cdiv(B * N1 * 256, 256), 256, 0, stream>>>(bufC, xfp, B * N1 * 256);

    // --- FP0: interpolate xfp (pos1 -> pts), concat pos, MLP 259->256->256 ---
    k_knn3<<<cdiv(B * N, 256), 256, 0, stream>>>(pos1, pts, N1, kidx, kw, N, B * N);
    k_interp<<<cdiv(B * N * 256, 256), 256, 0, stream>>>(
        xfp, kidx, kw, xup, N1, N, 256, B * N * 256);
    k_concat2_f16<<<cdiv(B * N * 288, 256), 256, 0, stream>>>(
        xup, 256, pts, 3, bufA, 288, B * N * 288);
    gemm(bufA, 22, bufB, B * N);
    gemm(bufB, 23, bufC, B * N);
    k_h2f<<<cdiv(B * N * 256, 256), 256, 0, stream>>>(bufC, (float*)d_out, B * N * 256);
}
